// LCGN_7344394076517
// MI455X (gfx1250) — matrix operand order repaired and checked by reading, not compiled, not run on hardware
//
#include <hip/hip_runtime.h>
#include <cstdint>
#include <cstddef>

// ---------------- types ----------------
typedef __bf16 bf16_t;
typedef __attribute__((ext_vector_type(16))) __bf16 v16bf;
typedef __attribute__((ext_vector_type(8)))  __bf16 v8bf;
typedef __attribute__((ext_vector_type(8)))  float  v8f;

#define NEG_BIG (-1e30f)

// problem constants
#define BB     128
#define NENT   100
#define LL     30
#define DFEAT  2112
#define CC     512
#define TT     4
#define MROWS  (BB * NENT)

// LDS row stride (halves): 32 data + 8 pad, keeps 16B alignment (80B rows)
#define SAPAD 40

// ---------------- WMMA helpers (CDNA5 wave32) ----------------
static __device__ __forceinline__ v8f wmma_bf16f32(v16bf a, v16bf b, v8f c) {
  // (neg_a, A, neg_b, B, c_mod, C, reuse_a, reuse_b)
  return __builtin_amdgcn_wmma_f32_16x16x32_bf16(false, a, false, b, (short)0, c,
                                                 false, false);
}

// A fragment 16x32 bf16 from LDS tile [rows][stride], K origin col 0.
// Lane layout (ISA 7.12.2): lanes 0-15 -> K {0..7,16..23}, lanes 16-31 -> K {8..15,24..31}
static __device__ __forceinline__ v16bf frag_a_lds(const bf16_t* s, int row_base, int stride) {
  const int lane = threadIdx.x & 31;
  const int r  = lane & 15;
  const int kb = (lane >> 4) << 3;  // 0 or 8
  const bf16_t* p = s + (row_base + r) * stride;
  v8bf lo = *(const v8bf*)(p + kb);
  v8bf hi = *(const v8bf*)(p + 16 + kb);
  v16bf o;
#pragma unroll
  for (int i = 0; i < 8; ++i) { o[i] = lo[i]; o[8 + i] = hi[i]; }
  return o;
}

// B fragment 32x16 (tile stored [cols][32 K] row-per-output-col).
// Lane n = lane&15; lanes 0-15 -> K {0..7,16..23}, lanes 16-31 -> K {8..15,24..31}.
static __device__ __forceinline__ v16bf frag_b_nt_lds(const bf16_t* s, int col_base, int stride) {
  const int lane = threadIdx.x & 31;
  const int n  = lane & 15;
  const int kb = (lane >> 4) << 3;  // 0 or 8
  const bf16_t* p = s + (col_base + n) * stride + kb;
  v8bf lo = *(const v8bf*)(p);
  v8bf hi = *(const v8bf*)(p + 16);
  v16bf o;
#pragma unroll
  for (int i = 0; i < 8; ++i) { o[i] = lo[i]; o[8 + i] = hi[i]; }
  return o;
}

// Async global -> LDS, 16 bytes per lane (ASYNCcnt path, no VGPR round-trip).
static __device__ __forceinline__ void async_g2l_b128(uint32_t lds_addr, const void* gaddr) {
  asm volatile("global_load_async_to_lds_b128 %0, %1, off"
               :: "v"(lds_addr), "v"(gaddr)
               : "memory");
}
static __device__ __forceinline__ void wait_async0() {
  asm volatile("s_wait_asynccnt 0" ::: "memory");
}
// low 32 bits of a generic LDS pointer == LDS byte offset (aperture mapping)
static __device__ __forceinline__ uint32_t lds_off32(const void* p) {
  return (uint32_t)(uintptr_t)p;
}

// ---------------- main NT GEMM: out = A[M,K] . W[N,K]^T + bias ----------------
// 256 threads = 8 waves (4 M x 2 N), wave tile 32x64, block tile 128x128, K step 32.
// Double-buffered LDS filled by async global->LDS loads (8 WMMA per 12 ds_load per wave-step).
__launch_bounds__(256)
__global__ void gemm_nt_main(const bf16_t* __restrict__ A, int lda,
                             const bf16_t* __restrict__ W, int ldw,
                             const float* __restrict__ bias,
                             float*  __restrict__ outf, int ldcf,
                             bf16_t* __restrict__ outh, int ldch,
                             const float* __restrict__ rowscale, int rowdiv, int nscale,
                             int K) {
  __shared__ __align__(16) bf16_t sA[2][128 * SAPAD];
  __shared__ __align__(16) bf16_t sW[2][128 * SAPAD];
  const int tid  = threadIdx.x;
  const int wave = tid >> 5;
  const int wm   = wave >> 1;   // 0..3 (M)
  const int wn   = wave & 1;    // 0..1 (N)
  const int m0   = blockIdx.x * 128;
  const int n0   = blockIdx.y * 128;
  v8f acc[2][4] = {};

  // stage one K-tile: A and W each 128 rows x 32 halves = 512 x 16B chunks, 2/thread, async DMA
  auto stage = [&](int buf, int k0) {
    const int r = tid >> 1, q = (tid & 1) * 2;
    const uint32_t la = lds_off32(&sA[buf][0]);
    async_g2l_b128(la + (uint32_t)(r * SAPAD + q * 8) * 2,
                   A + (size_t)(m0 + r) * lda + k0 + q * 8);
    async_g2l_b128(la + (uint32_t)(r * SAPAD + (q + 1) * 8) * 2,
                   A + (size_t)(m0 + r) * lda + k0 + (q + 1) * 8);
    const uint32_t lw = lds_off32(&sW[buf][0]);
    async_g2l_b128(lw + (uint32_t)(r * SAPAD + q * 8) * 2,
                   W + (size_t)(n0 + r) * ldw + k0 + q * 8);
    async_g2l_b128(lw + (uint32_t)(r * SAPAD + (q + 1) * 8) * 2,
                   W + (size_t)(n0 + r) * ldw + k0 + (q + 1) * 8);
  };

  stage(0, 0);
  wait_async0();
  __syncthreads();

  int buf = 0;
  for (int k0 = 0; k0 < K; k0 += 32) {
    if (k0 + 32 < K) stage(buf ^ 1, k0 + 32);   // prefetch next tile into other buffer

    const bf16_t* cA = &sA[buf][0];
    const bf16_t* cW = &sW[buf][0];
    v16bf a0 = frag_a_lds(cA, wm * 32,      SAPAD);
    v16bf a1 = frag_a_lds(cA, wm * 32 + 16, SAPAD);
    v16bf bfr[4];
#pragma unroll
    for (int j = 0; j < 4; ++j) bfr[j] = frag_b_nt_lds(cW, wn * 64 + j * 16, SAPAD);
#pragma unroll
    for (int j = 0; j < 4; ++j) {
      acc[0][j] = wmma_bf16f32(a0, bfr[j], acc[0][j]);
      acc[1][j] = wmma_bf16f32(a1, bfr[j], acc[1][j]);
    }

    wait_async0();        // this wave's async loads landed in LDS
    __syncthreads();      // everyone's landed; compute buffer free for reuse
    buf ^= 1;
  }

  // C/D layout: lane&15 = N, (lane>>4)*8 + vgpr = M within 16x16 tile
  const int lane = tid & 31;
  const int cn = lane & 15;
  const int rb = (lane >> 4) * 8;
#pragma unroll
  for (int i = 0; i < 2; ++i)
#pragma unroll
    for (int j = 0; j < 4; ++j) {
      const int mt = m0 + wm * 32 + i * 16 + rb;
      const int nt = n0 + wn * 64 + j * 16 + cn;
      const float bv = bias ? bias[nt] : 0.f;
#pragma unroll
      for (int v = 0; v < 8; ++v) {
        const int row = mt + v;
        const float val = acc[i][j][v] + bv;
        if (outf) outf[(size_t)row * ldcf + nt] = val;
        if (outh) {
          const float s = rowscale ? rowscale[(size_t)(row / rowdiv) * nscale + nt] : 1.f;
          outh[(size_t)row * ldch + nt] = (bf16_t)(val * s);
        }
      }
    }
}

// ---------------- batched, bounds-checked GEMM (attention) ----------------
// TRANSB=true : out = A[M,K] . B[N,K]^T (scores)
// TRANSB=false: out = A[M,K] . B[K,N]   (message) -- B transposed during staging
// 128 threads = 4 waves (2x2), block tile 64x64, K step 32.
template <bool TRANSB>
__launch_bounds__(128)
__global__ void gemm_batched_bf16(const bf16_t* __restrict__ A, int lda, long long strA,
                                  const bf16_t* __restrict__ Bm, int ldb, long long strB,
                                  float alpha,
                                  float*  __restrict__ outf, int ldcf, long long strCf,
                                  bf16_t* __restrict__ outh, int ldch, long long strCh,
                                  int M, int N, int K) {
  __shared__ __align__(16) bf16_t sA[64 * SAPAD];
  __shared__ __align__(16) bf16_t sB[64 * SAPAD];
  const int b = blockIdx.z;
  const bf16_t* Ab = A  + (size_t)b * strA;
  const bf16_t* Bb = Bm + (size_t)b * strB;
  const int tid  = threadIdx.x;
  const int wave = tid >> 5;
  const int wm   = wave >> 1;
  const int wn   = wave & 1;
  const int m0   = blockIdx.x * 64;
  const int n0   = blockIdx.y * 64;
  v8f acc[2][2] = {};

  for (int k0 = 0; k0 < K; k0 += 32) {
    for (int i = tid; i < 64 * 32; i += 128) {
      const int r = i >> 5, kk = i & 31;
      const int gr = m0 + r, gk = k0 + kk;
      sA[r * SAPAD + kk] = (gr < M && gk < K) ? Ab[(size_t)gr * lda + gk] : (bf16_t)0.f;
    }
    if (TRANSB) {
      for (int i = tid; i < 64 * 32; i += 128) {
        const int r = i >> 5, kk = i & 31;
        const int gn = n0 + r, gk = k0 + kk;
        sB[r * SAPAD + kk] = (gn < N && gk < K) ? Bb[(size_t)gn * ldb + gk] : (bf16_t)0.f;
      }
    } else {
      // transpose while staging: sB[col][k]
      for (int i = tid; i < 32 * 64; i += 128) {
        const int kk = i >> 6, c = i & 63;
        const int gk = k0 + kk, gn = n0 + c;
        sB[c * SAPAD + kk] = (gk < K && gn < N) ? Bb[(size_t)gk * ldb + gn] : (bf16_t)0.f;
      }
    }
    __syncthreads();

    v16bf a0 = frag_a_lds(sA, wm * 32,      SAPAD);
    v16bf a1 = frag_a_lds(sA, wm * 32 + 16, SAPAD);
    v16bf b0 = frag_b_nt_lds(sB, wn * 32,      SAPAD);
    v16bf b1 = frag_b_nt_lds(sB, wn * 32 + 16, SAPAD);
    acc[0][0] = wmma_bf16f32(a0, b0, acc[0][0]);
    acc[0][1] = wmma_bf16f32(a0, b1, acc[0][1]);
    acc[1][0] = wmma_bf16f32(a1, b0, acc[1][0]);
    acc[1][1] = wmma_bf16f32(a1, b1, acc[1][1]);
    __syncthreads();
  }

  const int lane = tid & 31;
  const int cn = lane & 15;
  const int rb = (lane >> 4) * 8;
#pragma unroll
  for (int i = 0; i < 2; ++i)
#pragma unroll
    for (int j = 0; j < 2; ++j) {
      const int mt = m0 + wm * 32 + i * 16 + rb;
      const int nt = n0 + wn * 32 + j * 16 + cn;
      if (nt >= N) continue;
#pragma unroll
      for (int v = 0; v < 8; ++v) {
        const int row = mt + v;
        if (row >= M) continue;
        const float val = acc[i][j][v] * alpha;
        if (outf) outf[(size_t)b * strCf + (size_t)row * ldcf + nt] = val;
        if (outh) outh[(size_t)b * strCh + (size_t)row * ldch + nt] = (bf16_t)val;
      }
    }
}

// ---------------- small fp32 / elementwise kernels ----------------
__global__ void f32_to_bf16_kernel(const float* __restrict__ x, bf16_t* __restrict__ y, int n) {
  const int i = blockIdx.x * blockDim.x + threadIdx.x;
  if (i < n) y[i] = (bf16_t)x[i];
}

// L2-normalize each row (K elems), write bf16
__global__ void row_norm_bf16_kernel(const float* __restrict__ x, bf16_t* __restrict__ y, int K) {
  __shared__ float red[256];
  const int row = blockIdx.x;
  const float* xr = x + (size_t)row * K;
  float ss = 0.f;
  for (int k = threadIdx.x; k < K; k += 256) { const float v = xr[k]; ss += v * v; }
  red[threadIdx.x] = ss;
  __syncthreads();
  for (int s = 128; s > 0; s >>= 1) {
    if (threadIdx.x < (unsigned)s) red[threadIdx.x] += red[threadIdx.x + s];
    __syncthreads();
  }
  const float rn = 1.f / fmaxf(sqrtf(red[0]), 1e-12f);
  bf16_t* yr = y + (size_t)row * K;
  for (int k = threadIdx.x; k < K; k += 256) yr[k] = (bf16_t)(xr[k] * rn);
}

__global__ void init_ctx_kernel(const float* __restrict__ initMem, bf16_t* __restrict__ xctx,
                                int total, int C) {
  const int i = blockIdx.x * blockDim.x + threadIdx.x;
  if (i < total) xctx[i] = (bf16_t)initMem[i % C];
}

__global__ void copy_rows_bf16_kernel(const bf16_t* __restrict__ src, int lds_,
                                      bf16_t* __restrict__ dst, int ldd, int rows, int cols) {
  const int i = blockIdx.x * blockDim.x + threadIdx.x;
  if (i >= rows * cols) return;
  const int r = i / cols, c = i % cols;
  dst[(size_t)r * ldd + c] = src[(size_t)r * lds_ + c];
}

__global__ void mul_f32_to_bf16_kernel(const float* __restrict__ a, const float* __restrict__ b,
                                       bf16_t* __restrict__ dst, int ldd, int coloff,
                                       int rows, int cols) {
  const int i = blockIdx.x * blockDim.x + threadIdx.x;
  if (i >= rows * cols) return;
  const int r = i / cols, c = i % cols;
  dst[(size_t)r * ldd + coloff + c] = (bf16_t)(a[i] * b[i]);
}

// out[m,n] = act( X[m,:].W[n,:] + bias[n] );  act: 0 none, 1 ELU
__global__ void small_linear_kernel(const float* __restrict__ X, const float* __restrict__ W,
                                    const float* __restrict__ bias, float* __restrict__ out,
                                    int M, int Nc, int K, int act) {
  const int i = blockIdx.x * blockDim.x + threadIdx.x;
  if (i >= M * Nc) return;
  const int m = i / Nc, n = i % Nc;
  float acc = bias ? bias[n] : 0.f;
  const float* xr = X + (size_t)m * K;
  const float* wr = W + (size_t)n * K;
  for (int k = 0; k < K; ++k) acc += xr[k] * wr[k];
  if (act == 1) acc = acc > 0.f ? acc : (expf(acc) - 1.f);
  out[i] = acc;
}

// Command attention: raw[l] = (q_cmd*lstm[l]).W_logit + b; masked softmax over L; cmd = att.lstm
__global__ void cmd_attend_kernel(const float* __restrict__ q_cmd, const float* __restrict__ lstm,
                                  const float* __restrict__ W_logit, const float* __restrict__ b_logit,
                                  const int* __restrict__ q_length, float* __restrict__ cmd,
                                  int L, int C) {
  __shared__ float att[64];
  const int b = blockIdx.x;
  const int tid = threadIdx.x;
  const int wave = tid >> 5, lane = tid & 31;
  for (int l = wave; l < L; l += 8) {
    float p = 0.f;
    const float* lr = lstm + ((size_t)b * L + l) * C;
    const float* qr = q_cmd + (size_t)b * C;
    for (int c = lane; c < C; c += 32) p += qr[c] * lr[c] * W_logit[c];
    for (int off = 16; off > 0; off >>= 1) p += __shfl_down(p, off, 32);
    if (lane == 0) att[l] = p + b_logit[0];
  }
  __syncthreads();
  if (tid < 32) {
    const int qlen = q_length[b];
    const bool on = (tid < L) && (tid < qlen);
    float v = on ? att[tid] : NEG_BIG;
    float mx = v;
    for (int off = 16; off > 0; off >>= 1) mx = fmaxf(mx, __shfl_xor(mx, off, 32));
    float e = on ? expf(v - mx) : 0.f;
    float s = e;
    for (int off = 16; off > 0; off >>= 1) s += __shfl_xor(s, off, 32);
    if (tid < L) att[tid] = e / s;
  }
  __syncthreads();
  for (int c = tid; c < C; c += 256) {
    float a = 0.f;
    for (int l = 0; l < L; ++l) a += att[l] * lstm[((size_t)b * L + l) * C + c];
    cmd[(size_t)b * C + c] = a;
  }
}

// Entity-masked softmax over k (Nk<=128), bf16 output
__global__ void attn_softmax_kernel(const float* __restrict__ scores, bf16_t* __restrict__ prob,
                                    const int* __restrict__ entity_num, int Nk) {
  __shared__ float red[128];
  const int row = blockIdx.x;        // b*Nk + q  (square attention)
  const int b = row / Nk;
  const int ne = entity_num[b];
  const float* sr = scores + (size_t)row * Nk;
  const int tid = threadIdx.x;
  const bool on = (tid < Nk) && (tid < ne);
  const float v = on ? sr[tid] : NEG_BIG;
  red[tid] = v;
  __syncthreads();
  for (int s = 64; s > 0; s >>= 1) {
    if (tid < s) red[tid] = fmaxf(red[tid], red[tid + s]);
    __syncthreads();
  }
  const float mx = red[0];
  __syncthreads();
  const float e = on ? expf(v - mx) : 0.f;
  red[tid] = e;
  __syncthreads();
  for (int s = 64; s > 0; s >>= 1) {
    if (tid < s) red[tid] += red[tid + s];
    __syncthreads();
  }
  const float inv = 1.f / red[0];
  if (tid < Nk) prob[(size_t)row * Nk + tid] = (bf16_t)(e * inv);
}

// ---------------- host orchestration ----------------
extern "C" void kernel_launch(void* const* d_in, const int* in_sizes, int n_in,
                              void* d_out, int out_size, void* d_ws, size_t ws_size,
                              hipStream_t stream) {
  (void)in_sizes; (void)n_in; (void)out_size; (void)ws_size;
  const float* images   = (const float*)d_in[0];
  const float* q_enc    = (const float*)d_in[1];
  const float* lstm     = (const float*)d_in[2];
  const float* W_initKB = (const float*)d_in[3];
  const float* b_initKB = (const float*)d_in[4];
  const float* initMem  = (const float*)d_in[5];
  const float* W_qIn    = (const float*)d_in[6];
  const float* b_qIn    = (const float*)d_in[7];
  const float* W_qIn2   = (const float*)d_in[8];
  const float* b_qIn2   = (const float*)d_in[9];
  const float* W_logit  = (const float*)d_in[10];
  const float* b_logit  = (const float*)d_in[11];
  const float* W_ploc   = (const float*)d_in[12];
  const float* b_ploc   = (const float*)d_in[13];
  const float* W_pctx   = (const float*)d_in[14];
  const float* b_pctx   = (const float*)d_in[15];
  const float* W_q      = (const float*)d_in[16];
  const float* b_q      = (const float*)d_in[17];
  const float* W_k      = (const float*)d_in[18];
  const float* b_k      = (const float*)d_in[19];
  const float* W_v      = (const float*)d_in[20];
  const float* b_v      = (const float*)d_in[21];
  const float* W_pk     = (const float*)d_in[22];
  const float* b_pk     = (const float*)d_in[23];
  const float* W_pv     = (const float*)d_in[24];
  const float* b_pv     = (const float*)d_in[25];
  const float* W_mem    = (const float*)d_in[26];
  const float* b_mem    = (const float*)d_in[27];
  const float* W_comb   = (const float*)d_in[28];
  const float* b_comb   = (const float*)d_in[29];
  const int* q_length   = (const int*)d_in[30];
  const int* entity_num = (const int*)d_in[31];
  float* out = (float*)d_out;

  // ---- workspace bump allocator ----
  char* base = (char*)d_ws;
  size_t off = 0;
  auto alloc = [&](size_t bytes) -> void* {
    off = (off + 255) & ~(size_t)255;
    void* p = base + off;
    off += bytes;
    return p;
  };

  bf16_t* A_img   = (bf16_t*)alloc((size_t)MROWS * DFEAT * 2);
  bf16_t* wb_init = (bf16_t*)alloc((size_t)CC * DFEAT * 2);
  bf16_t* wb_ploc = (bf16_t*)alloc((size_t)CC * CC * 2);
  bf16_t* wb_pctx = (bf16_t*)alloc((size_t)CC * CC * 2);
  bf16_t* wb_q    = (bf16_t*)alloc((size_t)CC * 3 * CC * 2);
  bf16_t* wb_k    = (bf16_t*)alloc((size_t)CC * 3 * CC * 2);
  bf16_t* wb_v    = (bf16_t*)alloc((size_t)CC * 3 * CC * 2);
  bf16_t* wb_mem  = (bf16_t*)alloc((size_t)CC * 2 * CC * 2);
  bf16_t* wb_comb = (bf16_t*)alloc((size_t)CC * 2 * CC * 2);
  bf16_t* x_loc   = (bf16_t*)alloc((size_t)MROWS * CC * 2);
  bf16_t* x_ctx   = (bf16_t*)alloc((size_t)MROWS * CC * 2);
  bf16_t* x_joint = (bf16_t*)alloc((size_t)MROWS * 3 * CC * 2);
  float*  proj_loc = (float*)alloc((size_t)MROWS * CC * 4);
  float*  proj_ctx = (float*)alloc((size_t)MROWS * CC * 4);
  bf16_t* x_mem_in = (bf16_t*)alloc((size_t)MROWS * 2 * CC * 2);
  float*  scoresb  = (float*)alloc((size_t)BB * NENT * NENT * 4);
  bf16_t* prob     = (bf16_t*)alloc((size_t)BB * NENT * NENT * 2);
  float*  q_cmd_base = (float*)alloc((size_t)BB * CC * 4);
  float*  q_cmd      = (float*)alloc((size_t)BB * CC * 4);
  float*  cmd        = (float*)alloc((size_t)BB * CC * 4);
  float*  pk_cmd     = (float*)alloc((size_t)BB * CC * 4);
  float*  pv_cmd     = (float*)alloc((size_t)BB * CC * 4);
  // transient aliases: Q/K/V live where proj_loc/proj_ctx were (already consumed)
  bf16_t* queries = (bf16_t*)proj_loc;
  bf16_t* keys    = (bf16_t*)proj_loc + (size_t)MROWS * CC;   // fits in f32 region
  bf16_t* vals    = (bf16_t*)proj_ctx;

  const int TPB = 256;
  auto cdiv = [](int a, int b) { return (a + b - 1) / b; };
  const float inv_sqrt_c = 0.04419417382415922f;  // 1/sqrt(512)

  // ---- weights -> bf16 ----
  f32_to_bf16_kernel<<<cdiv(CC * DFEAT, TPB), TPB, 0, stream>>>(W_initKB, wb_init, CC * DFEAT);
  f32_to_bf16_kernel<<<cdiv(CC * CC, TPB), TPB, 0, stream>>>(W_ploc, wb_ploc, CC * CC);
  f32_to_bf16_kernel<<<cdiv(CC * CC, TPB), TPB, 0, stream>>>(W_pctx, wb_pctx, CC * CC);
  f32_to_bf16_kernel<<<cdiv(CC * 3 * CC, TPB), TPB, 0, stream>>>(W_q, wb_q, CC * 3 * CC);
  f32_to_bf16_kernel<<<cdiv(CC * 3 * CC, TPB), TPB, 0, stream>>>(W_k, wb_k, CC * 3 * CC);
  f32_to_bf16_kernel<<<cdiv(CC * 3 * CC, TPB), TPB, 0, stream>>>(W_v, wb_v, CC * 3 * CC);
  f32_to_bf16_kernel<<<cdiv(CC * 2 * CC, TPB), TPB, 0, stream>>>(W_mem, wb_mem, CC * 2 * CC);
  f32_to_bf16_kernel<<<cdiv(CC * 2 * CC, TPB), TPB, 0, stream>>>(W_comb, wb_comb, CC * 2 * CC);

  // ---- stem ----
  row_norm_bf16_kernel<<<MROWS, 256, 0, stream>>>(images, A_img, DFEAT);
  gemm_nt_main<<<dim3(MROWS / 128, CC / 128), 256, 0, stream>>>(
      A_img, DFEAT, wb_init, DFEAT, b_initKB,
      nullptr, 0, x_loc, CC, nullptr, 1, CC, DFEAT);
  init_ctx_kernel<<<cdiv(MROWS * CC, TPB), TPB, 0, stream>>>(initMem, x_ctx, MROWS * CC, CC);
  small_linear_kernel<<<cdiv(BB * CC, TPB), TPB, 0, stream>>>(
      q_enc, W_qIn, b_qIn, q_cmd_base, BB, CC, CC, /*ELU*/ 1);
  // x_joint[:,0:C] = x_loc (constant across rounds)
  copy_rows_bf16_kernel<<<cdiv(MROWS * CC, TPB), TPB, 0, stream>>>(
      x_loc, CC, x_joint, 3 * CC, MROWS, CC);

  const dim3 gmain(MROWS / 128, CC / 128);
  for (int t = 0; t < TT; ++t) {
    // command pipeline (small, fp32)
    small_linear_kernel<<<cdiv(BB * CC, TPB), TPB, 0, stream>>>(
        q_cmd_base, W_qIn2 + (size_t)t * CC * CC, b_qIn2 + (size_t)t * CC, q_cmd, BB, CC, CC, 0);
    cmd_attend_kernel<<<BB, 256, 0, stream>>>(q_cmd, lstm, W_logit, b_logit, q_length, cmd, LL, CC);
    small_linear_kernel<<<cdiv(BB * CC, TPB), TPB, 0, stream>>>(cmd, W_pk, b_pk, pk_cmd, BB, CC, CC, 0);
    small_linear_kernel<<<cdiv(BB * CC, TPB), TPB, 0, stream>>>(cmd, W_pv, b_pv, pv_cmd, BB, CC, CC, 0);

    // x_joint[:,C:2C] = x_ctx
    copy_rows_bf16_kernel<<<cdiv(MROWS * CC, TPB), TPB, 0, stream>>>(
        x_ctx, CC, x_joint + CC, 3 * CC, MROWS, CC);
    // projections + fuse product into x_joint[:,2C:3C]
    gemm_nt_main<<<gmain, 256, 0, stream>>>(x_loc, CC, wb_ploc, CC, b_ploc,
                                            proj_loc, CC, nullptr, 0, nullptr, 1, CC, CC);
    gemm_nt_main<<<gmain, 256, 0, stream>>>(x_ctx, CC, wb_pctx, CC, b_pctx,
                                            proj_ctx, CC, nullptr, 0, nullptr, 1, CC, CC);
    mul_f32_to_bf16_kernel<<<cdiv(MROWS * CC, TPB), TPB, 0, stream>>>(
        proj_loc, proj_ctx, x_joint, 3 * CC, 2 * CC, MROWS, CC);

    // Q, K*pk(cmd), V*pv(cmd)   (scale fused into GEMM epilogue)
    gemm_nt_main<<<gmain, 256, 0, stream>>>(x_joint, 3 * CC, wb_q, 3 * CC, b_q,
                                            nullptr, 0, queries, CC, nullptr, 1, CC, 3 * CC);
    gemm_nt_main<<<gmain, 256, 0, stream>>>(x_joint, 3 * CC, wb_k, 3 * CC, b_k,
                                            nullptr, 0, keys, CC, pk_cmd, NENT, CC, 3 * CC);
    gemm_nt_main<<<gmain, 256, 0, stream>>>(x_joint, 3 * CC, wb_v, 3 * CC, b_v,
                                            nullptr, 0, vals, CC, pv_cmd, NENT, CC, 3 * CC);

    // scores = Q.K^T * 1/sqrt(C)  (batched NT, bounds-checked 100x100)
    gemm_batched_bf16<true><<<dim3(2, 2, BB), 128, 0, stream>>>(
        queries, CC, (long long)NENT * CC, keys, CC, (long long)NENT * CC, inv_sqrt_c,
        scoresb, NENT, (long long)NENT * NENT, nullptr, 0, 0, NENT, NENT, CC);
    attn_softmax_kernel<<<MROWS, 128, 0, stream>>>(scoresb, prob, entity_num, NENT);

    // memory update: concat([x_ctx, prob.V]) . W_mem^T + b
    copy_rows_bf16_kernel<<<cdiv(MROWS * CC, TPB), TPB, 0, stream>>>(
        x_ctx, CC, x_mem_in, 2 * CC, MROWS, CC);
    gemm_batched_bf16<false><<<dim3(2, 8, BB), 128, 0, stream>>>(
        prob, NENT, (long long)NENT * NENT, vals, CC, (long long)NENT * CC, 1.f,
        nullptr, 0, 0, x_mem_in + CC, 2 * CC, (long long)NENT * 2 * CC, NENT, CC, NENT);
    gemm_nt_main<<<gmain, 256, 0, stream>>>(x_mem_in, 2 * CC, wb_mem, 2 * CC, b_mem,
                                            nullptr, 0, x_ctx, CC, nullptr, 1, CC, 2 * CC);
  }

  // final: concat([x_loc, x_ctx]) . W_comb^T + b -> f32 out
  copy_rows_bf16_kernel<<<cdiv(MROWS * CC, TPB), TPB, 0, stream>>>(
      x_loc, CC, x_mem_in, 2 * CC, MROWS, CC);
  copy_rows_bf16_kernel<<<cdiv(MROWS * CC, TPB), TPB, 0, stream>>>(
      x_ctx, CC, x_mem_in + CC, 2 * CC, MROWS, CC);
  gemm_nt_main<<<gmain, 256, 0, stream>>>(x_mem_in, 2 * CC, wb_comb, 2 * CC, b_comb,
                                          out, CC, nullptr, 0, nullptr, 1, CC, 2 * CC);
}
